// Encoder_8194797601282
// MI455X (gfx1250) — matrix operand order repaired and checked by reading, not compiled, not run on hardware
//
#include <hip/hip_runtime.h>
#include <hip/hip_bf16.h>
#include <math.h>

typedef __attribute__((ext_vector_type(16))) __bf16 v16bf;
typedef __attribute__((ext_vector_type(8)))  float  v8f;

#define IN_CH 512
#define OUTC  256           // HEADS*HID
#define HEADS 4
#define NEG_SLOPE 0.2f

__device__ __forceinline__ float leaky(float v) { return v >= 0.f ? v : NEG_SLOPE * v; }

// ---------------------------------------------------------------------------
// Pack W (lin_w [256 x 512], row-major, h[n,o]=sum_k x[n,k]*W[o,k]) into
// bf16 hi/lo B-fragments laid out exactly as V_WMMA_F32_16X16X32_BF16 wants:
// frag[((kk*16+nt)*32 + lane)*16 + h] where lane: n=lane&15, k=kk*32+(lane>>4)*8+(h&7)+(h>>3)*16
// ---------------------------------------------------------------------------
__global__ __launch_bounds__(256) void pack_w_kernel(const float* __restrict__ w,
                                                     __bf16* __restrict__ whi,
                                                     __bf16* __restrict__ wlo) {
    int f = blockIdx.x * 256 + threadIdx.x;        // 0 .. 131071
    int hh = f & 15;
    int L  = (f >> 4) & 31;
    int nt = (f >> 9) & 15;
    int kk = f >> 13;
    int k   = kk * 32 + ((L >> 4) * 8) + (hh & 7) + ((hh >> 3) * 16);
    int col = nt * 16 + (L & 15);
    float v = w[col * IN_CH + k];
    __bf16 hi = (__bf16)v;
    whi[f] = hi;
    wlo[f] = (__bf16)(v - (float)hi);
}

// ---------------------------------------------------------------------------
// GEMM: h[N,256] = x[N,512] @ W^T, bf16x3 emulation on v_wmma_f32_16x16x32_bf16.
// One wave per 16-row tile, covering all 256 output columns (16 C tiles).
// ---------------------------------------------------------------------------
__global__ __launch_bounds__(256) void gemm_wmma_kernel(const float* __restrict__ x,
                                                        const __bf16* __restrict__ whi,
                                                        const __bf16* __restrict__ wlo,
                                                        float* __restrict__ h,
                                                        int n_tiles) {
    int wid  = threadIdx.x >> 5;
    int lane = threadIdx.x & 31;
    int tile = blockIdx.x * 8 + wid;
    if (tile >= n_tiles) return;

    int mlo   = lane & 15;
    int hi16  = lane >> 4;          // 0 or 1
    int row   = tile * 16 + mlo;    // A-row this lane supplies
    int kbase = hi16 * 8;

    v8f c[16];
    v8f cz = {0.f, 0.f, 0.f, 0.f, 0.f, 0.f, 0.f, 0.f};
#pragma unroll
    for (int i = 0; i < 16; ++i) c[i] = cz;

    const float* xrow = x + (size_t)row * IN_CH;

#pragma unroll 1
    for (int kk = 0; kk < IN_CH / 32; ++kk) {
        // A fragment: halves h=0..7 -> K = kk*32+kbase+h ; h=8..15 -> +16 more
        float av[16];
        const float* ap = xrow + kk * 32 + kbase;
#pragma unroll
        for (int t = 0; t < 8; ++t) av[t] = ap[t];
#pragma unroll
        for (int t = 0; t < 8; ++t) av[8 + t] = ap[16 + t];

        v16bf ahi, alo;
#pragma unroll
        for (int t = 0; t < 16; ++t) {
            __bf16 hi = (__bf16)av[t];
            ahi[t] = hi;
            alo[t] = (__bf16)(av[t] - (float)hi);
        }

        const __bf16* wb = whi + ((size_t)(kk * 16) * 32 + lane) * 16;
        const __bf16* wl = wlo + ((size_t)(kk * 16) * 32 + lane) * 16;
#pragma unroll
        for (int nt = 0; nt < 16; ++nt) {
            v16bf bhi = *(const v16bf*)(wb + (size_t)nt * 512);
            v16bf blo = *(const v16bf*)(wl + (size_t)nt * 512);
            c[nt] = __builtin_amdgcn_wmma_f32_16x16x32_bf16(false, ahi, false, bhi,
                                                            (short)0, c[nt], false, false);
            c[nt] = __builtin_amdgcn_wmma_f32_16x16x32_bf16(false, ahi, false, blo,
                                                            (short)0, c[nt], false, false);
            c[nt] = __builtin_amdgcn_wmma_f32_16x16x32_bf16(false, alo, false, bhi,
                                                            (short)0, c[nt], false, false);
        }
    }

    // C/D layout: lane n = lane&15 ; VGPR r holds m = r + (lane>>4)*8
#pragma unroll
    for (int nt = 0; nt < 16; ++nt) {
        int col = nt * 16 + mlo;
#pragma unroll
        for (int r = 0; r < 8; ++r) {
            int m = r + hi16 * 8;
            h[((size_t)(tile * 16 + m)) * OUTC + col] = c[nt][r];
        }
    }
}

// ---------------------------------------------------------------------------
// Per-node attention halves: a_src[n,h] = sum_c h[n,h,c]*att_src[h,c] (same dst)
// One wave per node; 8 contiguous channels per lane; 8-lane shuffle reduction.
// ---------------------------------------------------------------------------
__global__ __launch_bounds__(256) void att_kernel(const float* __restrict__ h,
                                                  const float* __restrict__ att_src,
                                                  const float* __restrict__ att_dst,
                                                  float* __restrict__ a_src,
                                                  float* __restrict__ a_dst, int n) {
    int wid = threadIdx.x >> 5, lane = threadIdx.x & 31;
    int node = blockIdx.x * 8 + wid;
    if (node >= n) return;
    const float* hp = h + (size_t)node * OUTC + lane * 8;
    float s = 0.f, d = 0.f;
#pragma unroll
    for (int j = 0; j < 8; ++j) {
        float v = hp[j];
        s += v * att_src[lane * 8 + j];
        d += v * att_dst[lane * 8 + j];
    }
    s += __shfl_xor(s, 1); d += __shfl_xor(d, 1);
    s += __shfl_xor(s, 2); d += __shfl_xor(d, 2);
    s += __shfl_xor(s, 4); d += __shfl_xor(d, 4);
    if ((lane & 7) == 0) {
        int head = lane >> 3;
        a_src[node * HEADS + head] = s;
        a_dst[node * HEADS + head] = d;
    }
}

// ---------------------------------------------------------------------------
// Init: out = 0, emax = -inf, denom = 0
// ---------------------------------------------------------------------------
__global__ __launch_bounds__(256) void init_kernel(float* __restrict__ emax,
                                                   float* __restrict__ denom,
                                                   float* __restrict__ out, int n) {
    int i = blockIdx.x * 256 + threadIdx.x;
    if (i < n * OUTC) out[i] = 0.f;
    if (i < n * HEADS) { emax[i] = -__builtin_inff(); denom[i] = 0.f; }
}

// ---------------------------------------------------------------------------
// Edge pass 1: segment max of leaky(a_src[src]+a_dst[dst]) over dst.
// Edge ids >= E are the implicit self-loops (src=dst=e-E).
// ---------------------------------------------------------------------------
__global__ __launch_bounds__(256) void edge_max_kernel(const int* __restrict__ ei,
                                                       const float* __restrict__ a_src,
                                                       const float* __restrict__ a_dst,
                                                       float* __restrict__ emax,
                                                       int E, int n) {
    int e = blockIdx.x * 256 + threadIdx.x;
    if (e >= E + n) return;
    int s, d;
    if (e < E) { s = ei[e]; d = ei[E + e]; } else { s = d = e - E; }
    float4 as = *(const float4*)(a_src + (size_t)s * 4);
    float4 ad = *(const float4*)(a_dst + (size_t)d * 4);
    float* ep = emax + (size_t)d * 4;
    __hip_atomic_fetch_max(ep + 0, leaky(as.x + ad.x), __ATOMIC_RELAXED, __HIP_MEMORY_SCOPE_AGENT);
    __hip_atomic_fetch_max(ep + 1, leaky(as.y + ad.y), __ATOMIC_RELAXED, __HIP_MEMORY_SCOPE_AGENT);
    __hip_atomic_fetch_max(ep + 2, leaky(as.z + ad.z), __ATOMIC_RELAXED, __HIP_MEMORY_SCOPE_AGENT);
    __hip_atomic_fetch_max(ep + 3, leaky(as.w + ad.w), __ATOMIC_RELAXED, __HIP_MEMORY_SCOPE_AGENT);
}

// ---------------------------------------------------------------------------
// Edge pass 2: denom = segment_sum(exp(e - emax[dst]))
// ---------------------------------------------------------------------------
__global__ __launch_bounds__(256) void edge_denom_kernel(const int* __restrict__ ei,
                                                         const float* __restrict__ a_src,
                                                         const float* __restrict__ a_dst,
                                                         const float* __restrict__ emax,
                                                         float* __restrict__ denom,
                                                         int E, int n) {
    int e = blockIdx.x * 256 + threadIdx.x;
    if (e >= E + n) return;
    int s, d;
    if (e < E) { s = ei[e]; d = ei[E + e]; } else { s = d = e - E; }
    float4 as = *(const float4*)(a_src + (size_t)s * 4);
    float4 ad = *(const float4*)(a_dst + (size_t)d * 4);
    float4 em = *(const float4*)(emax + (size_t)d * 4);
    float* dp = denom + (size_t)d * 4;
    unsafeAtomicAdd(dp + 0, __expf(leaky(as.x + ad.x) - em.x));
    unsafeAtomicAdd(dp + 1, __expf(leaky(as.y + ad.y) - em.y));
    unsafeAtomicAdd(dp + 2, __expf(leaky(as.z + ad.z) - em.z));
    unsafeAtomicAdd(dp + 3, __expf(leaky(as.w + ad.w) - em.w));
}

// ---------------------------------------------------------------------------
// Edge pass 3: out[dst,:] += alpha[head] * h[src,:]. One wave per edge,
// lane l handles channels l, l+32, ..., l+224 (coalesced; 8 f32 atomics/lane).
// ---------------------------------------------------------------------------
__global__ __launch_bounds__(256) void edge_scatter_kernel(const int* __restrict__ ei,
                                                           const float* __restrict__ a_src,
                                                           const float* __restrict__ a_dst,
                                                           const float* __restrict__ emax,
                                                           const float* __restrict__ denom,
                                                           const float* __restrict__ h,
                                                           float* __restrict__ out,
                                                           int E, int n) {
    int gid = blockIdx.x * 256 + threadIdx.x;
    int e = gid >> 5, lane = gid & 31;
    if (e >= E + n) return;
    int s, d;
    if (e < E) { s = ei[e]; d = ei[E + e]; } else { s = d = e - E; }
    float4 as = *(const float4*)(a_src + (size_t)s * 4);
    float4 ad = *(const float4*)(a_dst + (size_t)d * 4);
    float4 em = *(const float4*)(emax + (size_t)d * 4);
    float4 dn = *(const float4*)(denom + (size_t)d * 4);
    float4 al;
    al.x = __expf(leaky(as.x + ad.x) - em.x) / (dn.x + 1e-16f);
    al.y = __expf(leaky(as.y + ad.y) - em.y) / (dn.y + 1e-16f);
    al.z = __expf(leaky(as.z + ad.z) - em.z) / (dn.z + 1e-16f);
    al.w = __expf(leaky(as.w + ad.w) - em.w) / (dn.w + 1e-16f);

    const float* hs = h + (size_t)s * OUTC;
    float* od = out + (size_t)d * OUTC;
#pragma unroll
    for (int j = 0; j < 8; ++j) {
        int ch = lane + 32 * j;
        float a = (ch < 64) ? al.x : (ch < 128) ? al.y : (ch < 192) ? al.z : al.w;
        unsafeAtomicAdd(od + ch, hs[ch] * a);
    }
}

// ---------------------------------------------------------------------------
// Epilogue: out = PReLU(out + bias), per-channel prelu weight.
// ---------------------------------------------------------------------------
__global__ __launch_bounds__(256) void epilogue_kernel(float* __restrict__ out,
                                                       const float* __restrict__ bias,
                                                       const float* __restrict__ pw,
                                                       int total) {
    int i = blockIdx.x * 256 + threadIdx.x;
    if (i >= total) return;
    int c = i & (OUTC - 1);
    float v = out[i] + bias[c];
    out[i] = v >= 0.f ? v : pw[c] * v;
}

extern "C" void kernel_launch(void* const* d_in, const int* in_sizes, int n_in,
                              void* d_out, int out_size, void* d_ws, size_t ws_size,
                              hipStream_t stream) {
    const float* x        = (const float*)d_in[0];
    const int*   ei       = (const int*)  d_in[1];
    const float* lin_w    = (const float*)d_in[2];
    const float* att_src  = (const float*)d_in[3];
    const float* att_dst  = (const float*)d_in[4];
    const float* bias     = (const float*)d_in[5];
    const float* prelu_w  = (const float*)d_in[6];
    float* out = (float*)d_out;

    int n = in_sizes[0] / IN_CH;   // 50000
    int E = in_sizes[1] / 2;       // 800000

    // workspace carve-out
    float* h      = (float*)d_ws;                        // n*256
    float* a_src  = h      + (size_t)n * OUTC;           // n*4
    float* a_dst  = a_src  + (size_t)n * HEADS;          // n*4
    float* emax   = a_dst  + (size_t)n * HEADS;          // n*4
    float* denom  = emax   + (size_t)n * HEADS;          // n*4
    __bf16* whi   = (__bf16*)(denom + (size_t)n * HEADS);// 256*512
    __bf16* wlo   = whi + (size_t)OUTC * IN_CH;          // 256*512

    pack_w_kernel<<<(OUTC * IN_CH) / 256, 256, 0, stream>>>(lin_w, whi, wlo);

    int ntiles = (n + 15) / 16;
    gemm_wmma_kernel<<<(ntiles + 7) / 8, 256, 0, stream>>>(x, whi, wlo, h, ntiles);

    att_kernel<<<(n + 7) / 8, 256, 0, stream>>>(h, att_src, att_dst, a_src, a_dst, n);

    int total = n * OUTC;
    init_kernel<<<(total + 255) / 256, 256, 0, stream>>>(emax, denom, out, n);

    int etot = E + n;
    edge_max_kernel<<<(etot + 255) / 256, 256, 0, stream>>>(ei, a_src, a_dst, emax, E, n);
    edge_denom_kernel<<<(etot + 255) / 256, 256, 0, stream>>>(ei, a_src, a_dst, emax, denom, E, n);
    edge_scatter_kernel<<<((etot * 32) + 255) / 256, 256, 0, stream>>>(ei, a_src, a_dst, emax,
                                                                       denom, h, out, E, n);
    epilogue_kernel<<<(total + 255) / 256, 256, 0, stream>>>(out, bias, prelu_w, total);
}
